// BS_attention_67946382622889
// MI455X (gfx1250) — matrix operand order repaired and checked by reading, not compiled, not run on hardware
//
#include <hip/hip_runtime.h>
#include <hip/hip_bf16.h>
#include <math.h>

typedef _Float16 half_t;
typedef __attribute__((ext_vector_type(16))) _Float16 v16h;
typedef __attribute__((ext_vector_type(8)))  float    v8f;

union V16H { v16h v; half_t e[16]; };
union V8F  { v8f  v; float  e[8];  };

#define B_  16
#define C_  64
#define CH_ 32
#define H_  256
#define W_  256
#define HW_ (H_*W_)
#define K2_ 9

// Branch-free tanh: native V_TANH_F32 if the toolchain exposes it, else
// 1 - 2*rcp(exp(2x)+1) via v_exp_f32 + v_rcp_f32 (saturates correctly at +-1).
__device__ __forceinline__ float fast_tanh(float x) {
#if __has_builtin(__builtin_amdgcn_tanhf)
    return __builtin_amdgcn_tanhf(x);
#elif __has_builtin(__builtin_amdgcn_tanh_f32)
    return __builtin_amdgcn_tanh_f32(x);
#else
    float e = __expf(2.f * x);
    return 1.f - 2.f * __builtin_amdgcn_rcpf(e + 1.f);
#endif
}

// ---------------- workspace layout (bytes) ----------------
// x, x2 as f16 [B][64][H][W]; pooled stats and xh/xv as f32; packed A-tiles f16.
#define OFF_X    0ull
#define OFF_X2   (OFF_X  + (size_t)B_*C_*HW_*2)
#define OFF_PH   (OFF_X2 + (size_t)B_*C_*HW_*2)
#define OFF_PV   (OFF_PH + (size_t)B_*C_*W_*4)
#define OFF_XH   (OFF_PV + (size_t)B_*C_*H_*4)
#define OFF_XV   (OFF_XH + (size_t)B_*C_*W_*4)
#define OFF_AP   (OFF_XV + (size_t)B_*C_*H_*4)          // 32B aligned
// A-pack tile indices (each tile = 32 lanes x 16 halves = 512 halves)
#define T_SR 0      // 4 tiles  : sr_w  (cout tiles 0..3, K=32)
#define T_SP 4      // 8 tiles  : sp_w  (4 cout tiles x 2 K-steps)
#define T_RB 12     // 1 tile   : rb_w  (9 rows used)
#define T_LC 13     // 18 tiles : lc_w  (K = tap*64+cin, 18 K-steps)
#define T_CF 31     // 2 tiles  : cf_w
#define NTILES 33

// ISA 7.12.2 16-bit A-matrix 16x32 mapping: (lane, half-index i) -> (row m, k)
__device__ __forceinline__ void a_map(int lane, int i, int& m, int& k) {
    m = lane & 15;
    int bsel = lane >> 4, g = i >> 3, r = i & 7;
    k = g * 16 + bsel * 8 + r;
}

// ---------------- kernel 0: pack WMMA A-matrices (f32 -> f16, A-layout) ------
__global__ __launch_bounds__(256) void pack_weights(
    const float* __restrict__ sr_w, const float* __restrict__ sp_w,
    const float* __restrict__ rb_w, const float* __restrict__ lc_w,
    const float* __restrict__ cf_w, half_t* __restrict__ ap) {
    for (int idx = blockIdx.x * blockDim.x + threadIdx.x; idx < NTILES * 512;
         idx += gridDim.x * blockDim.x) {
        int tile = idx >> 9, within = idx & 511;
        int lane = within >> 4, i = within & 15;
        int m, k; a_map(lane, i, m, k);
        float val = 0.f;
        if (tile < T_SP) {                       // sr_w (64,32): cout=tile*16+m
            val = sr_w[(tile * 16 + m) * CH_ + k];
        } else if (tile < T_RB) {                // sp_w (64,64)
            int tt = (tile - T_SP) >> 1, s = (tile - T_SP) & 1;
            val = sp_w[(tt * 16 + m) * C_ + s * 32 + k];
        } else if (tile == T_RB) {               // rb_w (9,32), pad rows 9..15
            val = (m < K2_) ? rb_w[m * CH_ + k] : 0.f;
        } else if (tile < T_CF) {                // lc_w (9,64,3,3): K = t*64+cin
            int K = (tile - T_LC) * 32 + k;
            int t = K >> 6, cin = K & 63, ti = t / 3, tj = t % 3;
            val = (m < K2_) ? lc_w[((m * C_ + cin) * 3 + ti) * 3 + tj] : 0.f;
        } else {                                 // cf_w (9,64)
            int s = tile - T_CF;
            val = (m < K2_) ? cf_w[m * C_ + s * 32 + k] : 0.f;
        }
        ap[idx] = (half_t)val;
    }
}

// ---------------- kernel 1: x, x2, reblur (WMMA) -----------------------------
// One wave = one 16-pixel tile along W. B rows = input channels (lane&15 = pixel).
__global__ __launch_bounds__(256) void stage1(
    const float* __restrict__ feat,
    const float* __restrict__ sr_b, const float* __restrict__ sp_b,
    const float* __restrict__ rb_b, const half_t* __restrict__ ap,
    half_t* __restrict__ wx, half_t* __restrict__ wx2,
    float* __restrict__ out_reblur) {
    __shared__ __align__(32) half_t xls[8][C_][16];
    __shared__ __align__(32) half_t x2ls[8][C_][16];
    int lane = threadIdx.x & 31, wv = threadIdx.x >> 5;
    int tile = blockIdx.x * 8 + wv;
    int b = tile >> 12;              // H*W/16 = 4096 tiles per image
    int rem = tile & 4095;
    int h = rem >> 4;
    int w0 = (rem & 15) << 4;

    const float* fsh = feat + ((size_t)(b * C_) << 16) + (h << 8) + w0 + (lane & 15);        // sharp ch=k, px=lane&15
    const float* fbl = feat + ((size_t)(b * C_ + 32) << 16) + (h << 8) + w0 + (lane & 15);   // blur  ch=32+k
    V16H bsr, brb;
    #pragma unroll
    for (int n = 0; n < 16; ++n) { int kk = ((n >> 3) << 4) + ((lane >> 4) << 3) + (n & 7); bsr.e[n] = (half_t)fsh[(size_t)kk << 16]; brb.e[n] = (half_t)fbl[(size_t)kk << 16]; }

    // reblur_filter = tanh(rb_w @ blur + rb_b)  -> first output
    {
        v16h a = *(const v16h*)(ap + (T_RB << 9) + (lane << 4));
        v8f accv = {};
        accv = __builtin_amdgcn_wmma_f32_16x16x32_f16(false, a, false, brb.v, (short)0, accv, false, false);
        V8F acc; acc.v = accv;
        #pragma unroll
        for (int r = 0; r < 8; ++r) {
            int m = r + ((lane >> 4) << 3);
            int mc = (m < K2_) ? m : 0;                      // clamp bias index
            float tv = fast_tanh(acc.e[r] + rb_b[mc]);       // branch-free
            if (m < K2_)
                out_reblur[((size_t)(b * K2_ + m) << 16) + (h << 8) + w0 + (lane & 15)] = tv;
        }
    }

    // x = sr_w @ sharp + sr_b   (K=32, 4 cout tiles)
    #pragma unroll
    for (int t = 0; t < 4; ++t) {
        v16h a = *(const v16h*)(ap + ((T_SR + t) << 9) + (lane << 4));
        v8f accv = {};
        accv = __builtin_amdgcn_wmma_f32_16x16x32_f16(false, a, false, bsr.v, (short)0, accv, false, false);
        V8F acc; acc.v = accv;
        #pragma unroll
        for (int r = 0; r < 8; ++r) {
            int c = t * 16 + r + ((lane >> 4) << 3);
            xls[wv][c][lane & 15] = (half_t)(acc.e[r] + sr_b[c]);
        }
    }

    // restage D(f32)->B(f16) through LDS, then x2 = sp_w @ x + sp_b (K=64)
    V16H bx0u, bx1u;
    #pragma unroll
    for (int n = 0; n < 16; ++n) { int kk = ((n >> 3) << 4) + ((lane >> 4) << 3) + (n & 7); bx0u.e[n] = xls[wv][kk][lane & 15]; bx1u.e[n] = xls[wv][32 + kk][lane & 15]; }
    v16h bx0 = bx0u.v;
    v16h bx1 = bx1u.v;
    #pragma unroll
    for (int t = 0; t < 4; ++t) {
        v16h a0 = *(const v16h*)(ap + ((T_SP + t * 2 + 0) << 9) + (lane << 4));
        v16h a1 = *(const v16h*)(ap + ((T_SP + t * 2 + 1) << 9) + (lane << 4));
        v8f accv = {};
        accv = __builtin_amdgcn_wmma_f32_16x16x32_f16(false, a0, false, bx0, (short)0, accv, false, false);
        accv = __builtin_amdgcn_wmma_f32_16x16x32_f16(false, a1, false, bx1, (short)0, accv, false, false);
        V8F acc; acc.v = accv;
        #pragma unroll
        for (int r = 0; r < 8; ++r) {
            int c = t * 16 + r + ((lane >> 4) << 3);
            x2ls[wv][c][lane & 15] = (half_t)(acc.e[r] + sp_b[c]);
        }
    }

    // coalesced f16 tile writes: 2 channel-rows per lane, 2x16B each
    #pragma unroll
    for (int rr = 0; rr < 2; ++rr) {
        int c = lane + 32 * rr;
        size_t off = ((size_t)(b * C_ + c) << 16) + (h << 8) + w0;
        *(uint4*)(wx  + off)     = *(const uint4*)&xls[wv][c][0];
        *(uint4*)(wx  + off + 8) = *(const uint4*)&xls[wv][c][8];
        *(uint4*)(wx2 + off)     = *(const uint4*)&x2ls[wv][c][0];
        *(uint4*)(wx2 + off + 8) = *(const uint4*)&x2ls[wv][c][8];
    }
}

// ---------------- kernel 2: pooled means over H and W ------------------------
__global__ __launch_bounds__(256) void pool_means(
    const half_t* __restrict__ wx2, float* __restrict__ ph, float* __restrict__ pv) {
    int t = blockIdx.x * blockDim.x + threadIdx.x;
    const int N = B_ * C_ * W_;
    if (t < N) {                                   // ph[b,c,w] = mean over h
        int w = t & 255, c = (t >> 8) & 63, b = t >> 14;
        size_t base = ((size_t)(b * C_ + c) << 16) + w;
        float s = 0.f;
        for (int h = 0; h < H_; ++h) s += (float)wx2[base + ((size_t)h << 8)];
        ph[t] = s * (1.f / H_);
    } else {                                       // pv[b,c,h] = mean over w
        int t2 = t - N;
        int h = t2 & 255, c = (t2 >> 8) & 63, b = t2 >> 14;
        const half_t* row = wx2 + ((size_t)(b * C_ + c) << 16) + ((size_t)h << 8);
        float s = 0.f;
        for (int w = 0; w < W_; ++w) s += (float)row[w];
        pv[t2] = s * (1.f / W_);
    }
}

// ---------------- kernel 3: 1-D convs on pooled stats ------------------------
__global__ __launch_bounds__(256) void pooled_convs(
    const float* __restrict__ ch_w, const float* __restrict__ cv_w,
    const float* __restrict__ ph, const float* __restrict__ pv,
    float* __restrict__ xh, float* __restrict__ xv) {
    int t = blockIdx.x * blockDim.x + threadIdx.x;   // (b, m, i)
    int i = t & 255, m = (t >> 8) & 63, b = t >> 14;
    float sh = 0.f, sv = 0.f;
    for (int cin = 0; cin < C_; ++cin) {
        const float* phr = ph + ((b * C_ + cin) << 8);
        const float* pvr = pv + ((b * C_ + cin) << 8);
        const float* cwh = ch_w + (m * C_ + cin) * 3;
        const float* cwv = cv_w + (m * C_ + cin) * 3;
        #pragma unroll
        for (int d = 0; d < 3; ++d) {
            int j = i + d - 1;
            if ((unsigned)j < 256u) { sh += cwh[d] * phr[j]; sv += cwv[d] * pvr[j]; }
        }
    }
    xh[t] = sh; xv[t] = sv;
}

// ---------------- kernel 4: local_att, fusion, per-pixel convs, output -------
#define TW 64
__global__ __launch_bounds__(128) void stage3(
    const float* __restrict__ feat,
    const float* __restrict__ lc_b, const float* __restrict__ cf_b,
    const half_t* __restrict__ ap,
    const half_t* __restrict__ wx, const half_t* __restrict__ wx2,
    const float* __restrict__ xh, const float* __restrict__ xv,
    float* __restrict__ out2) {
    __shared__ half_t xt[C_][3][TW + 2];     // x  tile + halo
    __shared__ half_t x2t[C_][3][TW + 2];    // x2 tile + halo
    __shared__ float attL[4][K2_][16];       // tanh(local_att) per wave subtile
    __shared__ float attF[4][K2_][16];       // tanh(fusion)

    int blk = blockIdx.x;
    int w0 = (blk & 3) * TW;
    int h  = (blk >> 2) & 255;
    int b  = blk >> 10;

    // cooperative halo load (zero padded) -- fully uniform trip count
    const int NEL = C_ * 3 * (TW + 2);       // 12672 = 99 * 128
    for (int idx = threadIdx.x; idx < NEL; idx += 128) {
        int col = idx % (TW + 2);
        int rr  = (idx / (TW + 2)) % 3;
        int c   = idx / (3 * (TW + 2));
        int gh = h + rr - 1, gw = w0 + col - 1;
        half_t vx = (half_t)0.f, vx2 = (half_t)0.f;
        if ((unsigned)gh < (unsigned)H_ && (unsigned)gw < (unsigned)W_) {
            size_t off = ((size_t)(b * C_ + c) << 16) + (gh << 8) + gw;
            vx = wx[off]; vx2 = wx2[off];
        }
        xt[c][rr][col] = vx;
        x2t[c][rr][col] = vx2;
    }
    __syncthreads();

    int lane = threadIdx.x & 31, wv = threadIdx.x >> 5;
    int px0 = wv * 16;

    // local_att = tanh(conv3x3(x)): GEMM with K = 9 taps * 64 cin = 576
    {
        v8f accv = {};
        for (int step = 0; step < 18; ++step) {
            v16h a = *(const v16h*)(ap + ((T_LC + step) << 9) + (lane << 4));
            V16H bb;
            #pragma unroll
            for (int n = 0; n < 16; ++n) {
                int K = step * 32 + ((n >> 3) << 4) + ((lane >> 4) << 3) + (n & 7);
                int t = K >> 6, cin = K & 63, ti = t / 3, tj = t % 3;
                bb.e[n] = xt[cin][ti][px0 + tj + (lane & 15)];
            }
            accv = __builtin_amdgcn_wmma_f32_16x16x32_f16(false, a, false, bb.v, (short)0, accv, false, false);
        }
        V8F acc; acc.v = accv;
        #pragma unroll
        for (int r = 0; r < 8; ++r) {
            int m = r + ((lane >> 4) << 3);
            int mc = (m < K2_) ? m : 0;
            float tv = fast_tanh(acc.e[r] + lc_b[mc]);
            if (m < K2_) attL[wv][m][lane & 15] = tv;
        }
    }

    // fusion = tanh(cf_w @ leaky(xh + xv) + cf_b): K = 64, B built in registers
    {
        v8f accv = {};
        #pragma unroll
        for (int s = 0; s < 2; ++s) {
            v16h a = *(const v16h*)(ap + ((T_CF + s) << 9) + (lane << 4));
            V16H bb;
            #pragma unroll
            for (int n = 0; n < 16; ++n) {
                int k = s * 32 + ((n >> 3) << 4) + ((lane >> 4) << 3) + (n & 7);
                float v = xh[((size_t)(b * C_ + k) << 8) + w0 + px0 + (lane & 15)] + xv[((size_t)(b * C_ + k) << 8) + h];
                bb.e[n] = (half_t)(v > 0.f ? v : 0.2f * v);
            }
            accv = __builtin_amdgcn_wmma_f32_16x16x32_f16(false, a, false, bb.v, (short)0, accv, false, false);
        }
        V8F acc; acc.v = accv;
        #pragma unroll
        for (int r = 0; r < 8; ++r) {
            int m = r + ((lane >> 4) << 3);
            int mc = (m < K2_) ? m : 0;
            float tv = fast_tanh(acc.e[r] + cf_b[mc]);
            if (m < K2_) attF[wv][m][lane & 15] = tv;
        }
    }

    // per-pixel 3x3 weighted sums (VALU) + final subtract
    int px = lane & 15;
    int chalf = lane >> 4;
    int col = px0 + px;
    float aL[9], aF[9];
    #pragma unroll
    for (int t = 0; t < 9; ++t) { aL[t] = attL[wv][t][px]; aF[t] = attF[wv][t][px]; }
    for (int cc = 0; cc < 32; ++cc) {
        int c = cc + 32 * chalf;
        float sh = 0.f, lg = 0.f;
        #pragma unroll
        for (int t = 0; t < 9; ++t) {
            int ti = t / 3, tj = t % 3;
            sh += aL[t] * (float)xt[c][ti][col + tj];
            lg += aF[t] * (float)x2t[c][ti][col + tj];
        }
        float xc = (float)xt[c][1][col + 1];
        size_t g = ((size_t)(b * C_ + c) << 16) + ((size_t)h << 8) + (w0 + col);
        out2[g] = feat[g] - (xc * lg + sh);
    }
}

// ---------------- host launcher ----------------------------------------------
extern "C" void kernel_launch(void* const* d_in, const int* in_sizes, int n_in,
                              void* d_out, int out_size, void* d_ws, size_t ws_size,
                              hipStream_t stream) {
    const float* feat = (const float*)d_in[0];
    const float* sr_w = (const float*)d_in[1];
    const float* sr_b = (const float*)d_in[2];
    const float* sp_w = (const float*)d_in[3];
    const float* sp_b = (const float*)d_in[4];
    const float* ch_w = (const float*)d_in[5];
    const float* cv_w = (const float*)d_in[6];
    const float* cf_w = (const float*)d_in[7];
    const float* cf_b = (const float*)d_in[8];
    const float* lc_w = (const float*)d_in[9];
    const float* lc_b = (const float*)d_in[10];
    const float* rb_w = (const float*)d_in[11];
    const float* rb_b = (const float*)d_in[12];

    char* ws = (char*)d_ws;                 // requires ~272.7 MB of scratch
    half_t* wx  = (half_t*)(ws + OFF_X);
    half_t* wx2 = (half_t*)(ws + OFF_X2);
    float*  ph  = (float*)(ws + OFF_PH);
    float*  pv  = (float*)(ws + OFF_PV);
    float*  xh  = (float*)(ws + OFF_XH);
    float*  xv  = (float*)(ws + OFF_XV);
    half_t* apk = (half_t*)(ws + OFF_AP);

    float* out_reblur = (float*)d_out;
    float* out2 = (float*)d_out + (size_t)B_ * K2_ * HW_;

    pack_weights<<<66, 256, 0, stream>>>(sr_w, sp_w, rb_w, lc_w, cf_w, apk);
    stage1<<<(B_ * H_ * W_ / 16) / 8, 256, 0, stream>>>(
        feat, sr_b, sp_b, rb_b, apk, wx, wx2, out_reblur);
    pool_means<<<(2 * B_ * C_ * W_) / 256, 256, 0, stream>>>(wx2, ph, pv);
    pooled_convs<<<(B_ * C_ * W_) / 256, 256, 0, stream>>>(ch_w, cv_w, ph, pv, xh, xv);
    stage3<<<B_ * H_ * (W_ / TW), 128, 0, stream>>>(
        feat, lc_b, cf_b, apk, wx, wx2, xh, xv, out2);
}
